// WindowAttention_2121713844240
// MI455X (gfx1250) — matrix operand order repaired and checked by reading, not compiled, not run on hardware
//
#include <hip/hip_runtime.h>

// ---------------------------------------------------------------------------
// Fused window attention for MI455X (gfx1250, wave32, WMMA + TDM).
//   x:[4096,64,128]f32  w_qkv:[128,384]  b_qkv:[384]  w_proj:[128,128] b_proj:[128]
// One workgroup (8 wave32) per window; entire dataflow resident in LDS (160KB,
// 2 blocks/WGP). All four matmuls use v_wmma_f32_16x16x32_f16 (f16 in, f32 acc).
// Weights are converted+transposed to f16 once into d_ws, then DMA'd into LDS
// per block with tensor_load_to_lds (TDM), overlapped with x f32->f16 staging.
// ---------------------------------------------------------------------------

typedef __attribute__((ext_vector_type(16))) _Float16 v16h;
typedef __attribute__((ext_vector_type(8)))  _Float16 v8h;
typedef __attribute__((ext_vector_type(4)))  _Float16 v4h;
typedef __attribute__((ext_vector_type(8)))  float    v8f;
typedef __attribute__((ext_vector_type(4)))  unsigned int u32x4;
typedef __attribute__((ext_vector_type(8)))  int          i32x8;
typedef __attribute__((ext_vector_type(4)))  int          i32x4;

#define TOKENS 64
#define CDIM   128
#define QKVN   384
#define LDAT   65      // padded f32 attn row stride: bank = tid*65 mod 64 -> conflict-free

__device__ __forceinline__ v8f wmma_f16(v16h a, v16h b, v8f c) {
  return __builtin_amdgcn_wmma_f32_16x16x32_f16(false, a, false, b, (short)0, c,
                                                false, false);
}

// A fragment (16x32 f16, MxK) from row-major LDS tile (row stride ld halves).
__device__ __forceinline__ v16h load_a(const _Float16* __restrict__ s,
                                       int m0, int k0, int ld, int lane) {
  const _Float16* p = s + (m0 + (lane & 15)) * ld + k0 + ((lane >> 4) << 3);
  v8h lo = *(const v8h*)p;          // K = +0..7
  v8h hi = *(const v8h*)(p + 16);   // K = +16..23
  v16h r;
#pragma unroll
  for (int i = 0; i < 8; ++i) { r[i] = lo[i]; r[i + 8] = hi[i]; }
  return r;
}

// B fragment (32x16 f16, KxN) from a TRANSPOSED (N-major) LDS tile.
__device__ __forceinline__ v16h load_bT(const _Float16* __restrict__ s,
                                        int k0, int n0, int ld, int lane) {
  const _Float16* p = s + (n0 + (lane & 15)) * ld + k0 + ((lane >> 4) << 3);
  v8h lo = *(const v8h*)p;
  v8h hi = *(const v8h*)(p + 16);
  v16h r;
#pragma unroll
  for (int i = 0; i < 8; ++i) { r[i] = lo[i]; r[i + 8] = hi[i]; }
  return r;
}

// --- Tensor Data Mover: 1-D contiguous DMA global->LDS (n8 elements of 8B). ---
// D# per CDNA5 ISA ch.8: group0 {count=1, lds_addr, global_addr, type=2},
// group1 {data_size=3(8B), tensor_dim0=tile_dim0=n8, tensor_dim1=tile_dim1=1,
// tensor_dim0_stride=n8}, groups 2/3 zero (dims unused).
__device__ __forceinline__ void tdm_load_1d(const void* gptr, const void* lptr,
                                            unsigned n8) {
  unsigned long long ga = (unsigned long long)gptr;
  unsigned lds = (unsigned)(unsigned long long)lptr;   // low 32b of generic = LDS offset
  u32x4 g0;
  g0[0] = 1u;                                          // count=1 (valid user D#)
  g0[1] = lds;                                         // lds_addr
  g0[2] = (unsigned)(ga & 0xffffffffu);                // global_addr[31:0]
  g0[3] = (unsigned)((ga >> 32) & 0x01ffffffu) | (2u << 30); // addr[56:32] | type=2
  i32x8 g1;
  g1[0] = (int)(3u << 16);                             // data_size = 8B
  g1[1] = (int)((n8 & 0xffffu) << 16);                 // tensor_dim0[15:0]
  g1[2] = (int)((n8 >> 16) | (1u << 16));              // tensor_dim0[31:16] | tensor_dim1=1
  g1[3] = (int)((n8 & 0xffffu) << 16);                 // tensor_dim1 hi=0 | tile_dim0=n8
  g1[4] = 1;                                           // tile_dim1=1, tile_dim2=0
  g1[5] = (int)n8;                                     // tensor_dim0_stride[31:0]
  g1[6] = 0;
  g1[7] = 0;
  i32x4 z = {0, 0, 0, 0};
#if __clang_major__ >= 23
  i32x8 z8 = {0, 0, 0, 0, 0, 0, 0, 0};
  __builtin_amdgcn_tensor_load_to_lds(g0, g1, z, z, z8, 0);
#else
  __builtin_amdgcn_tensor_load_to_lds(g0, g1, z, z, 0);
#endif
}

// Phase-2 scratch overlays the (dead after GEMM1) 96KB W_qkv^T region.
struct Phase2 {
  _Float16 wprojT[CDIM * CDIM];      // 32 KB, N-major (transposed)
  float    attn [TOKENS * LDAT];     // 16.25 KB, padded rows
  _Float16 attnh[TOKENS * TOKENS];   //  8 KB
  _Float16 outh [TOKENS * CDIM];     // 16 KB
};
union WeightsU {
  _Float16 wqkvT[QKVN * CDIM];       // 96 KB, N-major (transposed)
  Phase2   p2;
};

// One-time weight convert+transpose f32 -> f16 into workspace.
__global__ void convert_weights_kernel(const float* __restrict__ w_qkv,
                                       const float* __restrict__ w_proj,
                                       _Float16* __restrict__ wqkvT,
                                       _Float16* __restrict__ wprojT) {
  int i = blockIdx.x * blockDim.x + threadIdx.x;
  if (i < CDIM * QKVN) {                       // wqkvT[n*128 + k] = w_qkv[k*384 + n]
    int n = i >> 7, k = i & 127;
    wqkvT[i] = (_Float16)w_qkv[k * QKVN + n];
  }
  int j = i - CDIM * QKVN;
  if (j >= 0 && j < CDIM * CDIM) {             // wprojT[n*128 + k] = w_proj[k*128 + n]
    int n = j >> 7, k = j & 127;
    wprojT[j] = (_Float16)w_proj[k * CDIM + n];
  }
}

template <bool USE_WS>
__global__ __launch_bounds__(256, 1)
void win_attn_kernel(const float* __restrict__ x,
                     const float* __restrict__ b_qkv,
                     const float* __restrict__ b_proj,
                     float* __restrict__ out,
                     const _Float16* __restrict__ wqkvT_g,
                     const _Float16* __restrict__ wprojT_g,
                     const float* __restrict__ wqkv_f,
                     const float* __restrict__ wproj_f) {
  __shared__ __align__(32) _Float16 s_xh[TOKENS * CDIM];   // 16 KB
  __shared__ __align__(32) _Float16 s_q [TOKENS * CDIM];   // 16 KB
  __shared__ __align__(32) _Float16 s_k [TOKENS * CDIM];   // 16 KB
  __shared__ __align__(32) _Float16 s_vT[CDIM * TOKENS];   // 16 KB (transposed [c][m])
  __shared__ __align__(32) WeightsU s_u;                   // 96 KB  -> 160 KB total

  const int tid  = threadIdx.x;
  const int lane = tid & 31;
  const int wave = tid >> 5;
  const size_t blk = blockIdx.x;

  // Every wave owns M-tile (wave&3)*16 in every GEMM; n tiles step by 32*j.
  const int m0    = (wave & 3) * 16;
  const int nbase = (wave >> 2) * 16;
  const int mb    = m0 + ((lane >> 4) << 3);   // D-frag row base for this lane
  const int nlane = lane & 15;

  // ---------------- phase 1: stage x (f32->f16); DMA W_qkv^T into LDS -------
  if (USE_WS) {
    if (wave == 0) tdm_load_1d(wqkvT_g, s_u.wqkvT, (QKVN * CDIM) / 4);  // 12288 x 8B
  } else {
    for (int i = tid; i < CDIM * QKVN; i += 256) {
      int k = i / QKVN, n = i - k * QKVN;
      s_u.wqkvT[n * CDIM + k] = (_Float16)wqkv_f[i];
    }
  }
  {
    const float4* xs = (const float4*)(x + blk * (TOKENS * CDIM));
    v4h* xd = (v4h*)s_xh;
#pragma unroll 2
    for (int i = tid; i < (TOKENS * CDIM) / 4; i += 256) {
      float4 t = xs[i];
      v4h h; h[0] = (_Float16)t.x; h[1] = (_Float16)t.y;
             h[2] = (_Float16)t.z; h[3] = (_Float16)t.w;
      xd[i] = h;
    }
  }
  if (USE_WS && wave == 0) __builtin_amdgcn_s_wait_tensorcnt(0);
  __syncthreads();

  // ---------------- phase 2: GEMM1  qkv = x @ W_qkv + b ----------------------
  // 12 n-tiles per wave: n0 = nbase + 32*j; j<4 -> q, j<8 -> k, else -> v^T.
  {
    v16h xa[4];
#pragma unroll
    for (int i = 0; i < 4; ++i) xa[i] = load_a(s_xh, m0, i * 32, CDIM, lane);
    float biasv[12];
#pragma unroll
    for (int j = 0; j < 12; ++j) biasv[j] = b_qkv[nbase + 32 * j + nlane];

#pragma unroll
    for (int j = 0; j < 12; ++j) {
      const int n0 = nbase + 32 * j;
      v16h bf[4];
#pragma unroll
      for (int i = 0; i < 4; ++i) bf[i] = load_bT(s_u.wqkvT, i * 32, n0, CDIM, lane);
      v8f c;
#pragma unroll
      for (int r = 0; r < 8; ++r) c[r] = biasv[j];
#pragma unroll
      for (int i = 0; i < 4; ++i) c = wmma_f16(xa[i], bf[i], c);

      const int nn = n0 + nlane;
      if (j < 4) {                       // q tile (n0 < 128)
#pragma unroll
        for (int r = 0; r < 8; ++r) s_q[(mb + r) * CDIM + nn] = (_Float16)c[r];
      } else if (j < 8) {                // k tile
        const int n2 = nn - CDIM;
#pragma unroll
        for (int r = 0; r < 8; ++r) s_k[(mb + r) * CDIM + n2] = (_Float16)c[r];
      } else {                           // v tile, stored transposed [c][m]
        const int n2 = nn - 2 * CDIM;
#pragma unroll
        for (int r = 0; r < 8; ++r) s_vT[n2 * TOKENS + (mb + r)] = (_Float16)c[r];
      }
    }
  }
  __syncthreads();

  // ---------------- phase 3: DMA W_proj^T; attn = q @ k^T * scale ------------
  if (USE_WS) {
    if (wave == 0) tdm_load_1d(wprojT_g, s_u.p2.wprojT, (CDIM * CDIM) / 4); // 4096 x 8B
  } else {
    for (int i = tid; i < CDIM * CDIM; i += 256) {
      int k = i >> 7, n = i & 127;
      s_u.p2.wprojT[n * CDIM + k] = (_Float16)wproj_f[i];
    }
  }
  {
    const float scale = 0.08838834764831845f;   // 128^-0.5
    v16h qa[4];
#pragma unroll
    for (int i = 0; i < 4; ++i) qa[i] = load_a(s_q, m0, i * 32, CDIM, lane);
#pragma unroll
    for (int j = 0; j < 2; ++j) {               // 2 n-tiles per wave over [64x64]
      const int n0 = nbase + 32 * j;
      v16h bf[4];
#pragma unroll
      for (int i = 0; i < 4; ++i) bf[i] = load_bT(s_k, i * 32, n0, CDIM, lane);
      v8f c = {};
#pragma unroll
      for (int i = 0; i < 4; ++i) c = wmma_f16(qa[i], bf[i], c);
      const int nn = n0 + nlane;
#pragma unroll
      for (int r = 0; r < 8; ++r) s_u.p2.attn[(mb + r) * LDAT + nn] = c[r] * scale;
    }
  }
  if (USE_WS && wave == 0) __builtin_amdgcn_s_wait_tensorcnt(0);
  __syncthreads();

  // ---------------- phase 4: row softmax (64 rows, one thread each) ----------
  if (tid < TOKENS) {
    float* row = s_u.p2.attn + tid * LDAT;
    float mx = row[0];
#pragma unroll 4
    for (int j = 1; j < TOKENS; ++j) mx = fmaxf(mx, row[j]);
    float ssum = 0.f;
#pragma unroll 4
    for (int j = 0; j < TOKENS; ++j) { float e = __expf(row[j] - mx); row[j] = e; ssum += e; }
    const float rs = 1.0f / ssum;
    _Float16* rh = s_u.p2.attnh + tid * TOKENS;
#pragma unroll 4
    for (int j = 0; j < TOKENS; ++j) rh[j] = (_Float16)(row[j] * rs);
  }
  __syncthreads();

  // ---------------- phase 5: out = attn @ v ----------------------------------
  {
    v16h aa[2];
#pragma unroll
    for (int i = 0; i < 2; ++i) aa[i] = load_a(s_u.p2.attnh, m0, i * 32, TOKENS, lane);
#pragma unroll
    for (int j = 0; j < 4; ++j) {               // 4 n-tiles per wave over [64x128]
      const int n0 = nbase + 32 * j;
      v16h bf[2];
#pragma unroll
      for (int i = 0; i < 2; ++i) bf[i] = load_bT(s_vT, i * 32, n0, TOKENS, lane);
      v8f c = {};
#pragma unroll
      for (int i = 0; i < 2; ++i) c = wmma_f16(aa[i], bf[i], c);
      const int nn = n0 + nlane;
#pragma unroll
      for (int r = 0; r < 8; ++r) s_u.p2.outh[(mb + r) * CDIM + nn] = (_Float16)c[r];
    }
  }
  __syncthreads();

  // ---------------- phase 6: y = out @ W_proj + b -> global -------------------
  {
    float* outp = out + blk * (TOKENS * CDIM);
    v16h oa[4];
#pragma unroll
    for (int i = 0; i < 4; ++i) oa[i] = load_a(s_u.p2.outh, m0, i * 32, CDIM, lane);
    float biasv[4];
#pragma unroll
    for (int j = 0; j < 4; ++j) biasv[j] = b_proj[nbase + 32 * j + nlane];
#pragma unroll
    for (int j = 0; j < 4; ++j) {
      const int n0 = nbase + 32 * j;
      v16h bf[4];
#pragma unroll
      for (int i = 0; i < 4; ++i) bf[i] = load_bT(s_u.p2.wprojT, i * 32, n0, CDIM, lane);
      v8f c;
#pragma unroll
      for (int r = 0; r < 8; ++r) c[r] = biasv[j];
#pragma unroll
      for (int i = 0; i < 4; ++i) c = wmma_f16(oa[i], bf[i], c);
      const int nn = n0 + nlane;
#pragma unroll
      for (int r = 0; r < 8; ++r) outp[(mb + r) * CDIM + nn] = c[r];
    }
  }
}

extern "C" void kernel_launch(void* const* d_in, const int* in_sizes, int n_in,
                              void* d_out, int out_size, void* d_ws, size_t ws_size,
                              hipStream_t stream) {
  const float* x      = (const float*)d_in[0];
  const float* w_qkv  = (const float*)d_in[1];
  const float* b_qkv  = (const float*)d_in[2];
  const float* w_proj = (const float*)d_in[3];
  const float* b_proj = (const float*)d_in[4];
  float* out = (float*)d_out;
  const int B = in_sizes[0] / (TOKENS * CDIM);   // 4096 windows

  const size_t WS_NEED = (size_t)(QKVN * CDIM + CDIM * CDIM) * sizeof(_Float16);
  if (d_ws != nullptr && ws_size >= WS_NEED) {
    _Float16* wqkvT  = (_Float16*)d_ws;
    _Float16* wprojT = wqkvT + QKVN * CDIM;
    const int tot = CDIM * QKVN + CDIM * CDIM;
    convert_weights_kernel<<<(tot + 255) / 256, 256, 0, stream>>>(w_qkv, w_proj,
                                                                  wqkvT, wprojT);
    win_attn_kernel<true><<<B, 256, 0, stream>>>(x, b_qkv, b_proj, out,
                                                 wqkvT, wprojT, w_qkv, w_proj);
  } else {
    win_attn_kernel<false><<<B, 256, 0, stream>>>(x, b_qkv, b_proj, out,
                                                  nullptr, nullptr, w_qkv, w_proj);
  }
}
